// mha_59751585022401
// MI455X (gfx1250) — matrix operand order repaired and checked by reading, not compiled, not run on hardware
//
#include <hip/hip_runtime.h>
#include <hip/hip_bf16.h>

#define BB 4
#define SS 1024
#define HD 1024
#define NH 16
#define NEG_BIG (-4294967295.0f)

typedef __bf16 v16bf __attribute__((ext_vector_type(16)));
typedef __bf16 v8bf  __attribute__((ext_vector_type(8)));
typedef __bf16 v4bf  __attribute__((ext_vector_type(4)));
typedef float  v8f   __attribute__((ext_vector_type(8)));
typedef unsigned int v4u __attribute__((ext_vector_type(4)));
typedef int v8i __attribute__((ext_vector_type(8)));
typedef int v4i __attribute__((ext_vector_type(4)));

__device__ __forceinline__ v8f vzero8() {
    v8f z = {0.f,0.f,0.f,0.f,0.f,0.f,0.f,0.f};
    return z;
}

__device__ __forceinline__ v8f wmma_bf16(v16bf a, v16bf b, v8f c) {
    // (neg_a, A, neg_b, B, c_mod, C, reuse_a, reuse_b)
    return __builtin_amdgcn_wmma_f32_16x16x32_bf16(false, a, false, b, (short)0, c, false, false);
}

__device__ __forceinline__ v8bf ld8(const __bf16* p) {
    return *(const v8bf*)p;                  // 16B aligned by construction -> b128
}

__device__ __forceinline__ v16bf join8(v8bf lo, v8bf hi) {
    return __builtin_shufflevector(lo, hi, 0,1,2,3,4,5,6,7,8,9,10,11,12,13,14,15);
}

// ---------------------------------------------------------------------------
// TDM: 1-D tile load (contiguous `n2` bf16 elements) global -> LDS.
// D# per CDNA5 ISA: group0 {count=1, lds_addr,
// global_addr, type=2}; group1 {data_size=1(2B), tensor_dim0=tile_dim0=n2,
// tensor_dim1=tile_dim1=1, stride0=n2}. Groups 2/3 zero (<=2D tensor).
// ---------------------------------------------------------------------------
__device__ __forceinline__ void tdm_load_1d(const void* gptr, unsigned lds_off,
                                            unsigned n2 /* bf16 elems, <=65535 */) {
    const unsigned long long ga = (unsigned long long)gptr;
    v4u g0;
    g0[0] = 1u;                                           // count=1 (valid user D#)
    g0[1] = lds_off;                                      // LDS byte address
    g0[2] = (unsigned)(ga & 0xffffffffu);                 // global_addr[31:0]
    g0[3] = (unsigned)((ga >> 32) & 0x01ffffffu) | (2u << 30);  // [56:32] | type=2
    v8i g1;
    g1[0] = (int)(1u << 16);                              // data_size=1 -> 2 bytes
    g1[1] = (int)(n2 << 16);                              // tensor_dim0[15:0]
    g1[2] = (int)(((n2 >> 16) & 0xffffu) | (1u << 16));   // tdim0[31:16] | tensor_dim1=1
    g1[3] = (int)((n2 & 0xffffu) << 16);                  // tile_dim0 = n2
    g1[4] = 1;                                            // tile_dim1=1, tile_dim2=0
    g1[5] = (int)n2;                                      // tensor_dim0_stride
    g1[6] = 0;
    g1[7] = 0;
    v4i z4 = {0, 0, 0, 0};
#if defined(__clang_major__) && __clang_major__ >= 23
    v8i z8 = {0, 0, 0, 0, 0, 0, 0, 0};
    __builtin_amdgcn_tensor_load_to_lds(g0, g1, z4, z4, z8, 0);
#else
    __builtin_amdgcn_tensor_load_to_lds(g0, g1, z4, z4, 0);
#endif
}

// ---------------------------------------------------------------------------
// Kernel 0: one-time f32 -> bf16 conversion (4 floats per thread)
// ---------------------------------------------------------------------------
__global__ __launch_bounds__(256) void cvt_kernel(const float* __restrict__ src,
                                                  __bf16* __restrict__ dst) {
    const int i = blockIdx.x * 256 + threadIdx.x;
    const float4 f = ((const float4*)src)[i];
    v4bf o = {(__bf16)f.x, (__bf16)f.y, (__bf16)f.z, (__bf16)f.w};
    ((v4bf*)dst)[i] = o;
}

// ---------------------------------------------------------------------------
// Kernel 1: row masks  k_mask = sign(sum|k|), q_mask = sign(sum|q|)
// ---------------------------------------------------------------------------
__global__ __launch_bounds__(256) void mask_kernel(const float* __restrict__ q,
                                                   const float* __restrict__ k,
                                                   float* __restrict__ qmask,
                                                   float* __restrict__ kmask) {
    __shared__ float rq[256], rk[256];
    const int row = blockIdx.x;
    const int t = threadIdx.x;
    const float* qr = q + (size_t)row * HD;
    const float* kr = k + (size_t)row * HD;
    float sq = 0.f, sk = 0.f;
    for (int j = t; j < HD; j += 256) { sq += fabsf(qr[j]); sk += fabsf(kr[j]); }
    rq[t] = sq; rk[t] = sk;
    __syncthreads();
    for (int s = 128; s > 0; s >>= 1) {
        if (t < s) { rq[t] += rq[t + s]; rk[t] += rk[t + s]; }
        __syncthreads();
    }
    if (t == 0) {
        qmask[row] = (rq[0] > 0.f) ? 1.f : 0.f;
        kmask[row] = (rk[0] > 0.f) ? 1.f : 0.f;
    }
}

// ---------------------------------------------------------------------------
// Kernel 2: fused QKV projection  Y = relu(X @ W^T + b), all-bf16.
// A tile staged once per block via TDM. K loop FULLY UNROLLED: every
// fragment is a fresh SSA value -> no cross-iteration register copies,
// scheduler hoists global/ds loads across the 128 WMMAs.
// ---------------------------------------------------------------------------
__global__ __launch_bounds__(256) void qkv_proj_kernel(const __bf16* __restrict__ xq,
                                                       const __bf16* __restrict__ xk,
                                                       const __bf16* __restrict__ xv,
                                                       const __bf16* __restrict__ W16,
                                                       const float* __restrict__ bias,
                                                       __bf16* __restrict__ q1,
                                                       __bf16* __restrict__ k1,
                                                       __bf16* __restrict__ v1T) {
    __shared__ __align__(16) __bf16 atile[16][HD];   // 32 KB block-shared A tile

    const int wave = threadIdx.x >> 5;
    const int wid  = blockIdx.x * 8 + wave;
    const int lane = threadIdx.x & 31;
    const int ln16 = lane & 15;
    const int half = lane >> 4;

    const int nGroup = wid & 15;           // 16 groups of 64 cols (== head id)
    const int mTile  = (wid >> 4) & 255;   // uniform across block
    const int mat    = wid >> 12;          // uniform across block: 0:q 1:k 2:v

    const __bf16* X = (mat == 0) ? xq : (mat == 1) ? xk : xv;

    // --- TDM: stage the block's 16x1024 bf16 A tile (contiguous 32 KB) ---
    if (wave == 0) {
        const unsigned lds_off = (unsigned)(unsigned long long)(void*)&atile[0][0];
        tdm_load_1d(X + (size_t)mTile * 16 * HD, lds_off, 16 * HD);
        __builtin_amdgcn_s_wait_tensorcnt(0);
    }
    __syncthreads();

    const __bf16* brow[4];
#pragma unroll
    for (int ns = 0; ns < 4; ++ns)
        brow[ns] = W16 + (size_t)(nGroup * 64 + ns * 16 + ln16) * HD;

    v8f acc[4] = {vzero8(), vzero8(), vzero8(), vzero8()};

#pragma unroll
    for (int kt = 0; kt < HD / 32; ++kt) {
        const int kBase = kt * 32;
        if ((kt & 7) == 0)                                   // one prefetch per 8 steps
            __builtin_prefetch(brow[0] + kBase + 1024, 0, 3);
        v16bf b[4];
#pragma unroll
        for (int ns = 0; ns < 4; ++ns)
            b[ns] = join8(ld8(brow[ns] + kBase + half * 8),
                          ld8(brow[ns] + kBase + 16 + half * 8));
        v16bf a = join8(ld8(&atile[ln16][kBase + half * 8]),     // ds_load_b128
                        ld8(&atile[ln16][kBase + 16 + half * 8]));
#pragma unroll
        for (int ns = 0; ns < 4; ++ns)
            acc[ns] = wmma_bf16(a, b[ns], acc[ns]);
    }

    if (mat < 2) {
        __bf16* Y = (mat == 0) ? q1 : k1;
#pragma unroll
        for (int ns = 0; ns < 4; ++ns) {
            const int n = nGroup * 64 + ns * 16 + ln16;
            const float bv = bias[n];
#pragma unroll
            for (int r = 0; r < 8; ++r) {
                const int row = mTile * 16 + r + half * 8;
                float val = acc[ns][r] + bv;
                Y[(size_t)row * HD + n] = (__bf16)(val > 0.f ? val : 0.f);
            }
        }
    } else {
        // v -> transposed layout: v1T[((h*BB + b)*64 + f)*SS + s], packed b128 stores
        const int h = nGroup;
        const int bbatch = mTile >> 6;
        const int sBase = (mTile & 63) * 16 + half * 8;
#pragma unroll
        for (int ns = 0; ns < 4; ++ns) {
            const int f = ns * 16 + ln16;
            const float bv = bias[h * 64 + f];
            v8bf ov;
#pragma unroll
            for (int r = 0; r < 8; ++r) {
                float val = acc[ns][r] + bv;
                ov[r] = (__bf16)(val > 0.f ? val : 0.f);
            }
            __bf16* dst = v1T + ((size_t)((h * BB + bbatch) * 64 + f)) * SS + sBase;
            *(v8bf*)dst = ov;
        }
    }
}

// ---------------------------------------------------------------------------
// Kernel 3: attention for one (batch, head, 16-query tile), all-bf16 operands.
// Scores and PV loops fully unrolled for load/WMMA overlap without copies.
// ---------------------------------------------------------------------------
__global__ __launch_bounds__(256) void attention_kernel(const __bf16* __restrict__ q1,
                                                        const __bf16* __restrict__ k1,
                                                        const __bf16* __restrict__ v1T,
                                                        const float* __restrict__ kmask,
                                                        const float* __restrict__ qmask,
                                                        const int* __restrict__ maskedp,
                                                        float* __restrict__ out) {
    __shared__ __align__(16) __bf16 sc[16][SS];   // 32 KB probs tile (WMMA-ready)
    __shared__ float red[16][16];
    __shared__ float pacc[16][64];

    const int qt = blockIdx.x & 63;
    const int h  = (blockIdx.x >> 6) & 15;
    const int bb = blockIdx.x >> 10;
    const int t    = threadIdx.x;
    const int wave = t >> 5, lane = t & 31, ln16 = lane & 15, half = lane >> 4;
    const int masked = *maskedp;

    for (int i = t; i < 1024; i += 256) ((float*)pacc)[i] = 0.f;

    // --- Q A-fragments (16 rows x 64 feats), reused for all 64 key chunks ---
    const int qrow = qt * 16 + ln16;
    const __bf16* qp = q1 + ((size_t)(bb * SS + qrow)) * HD + h * 64;
    v16bf a0 = join8(ld8(qp + half * 8),      ld8(qp + 16 + half * 8));
    v16bf a1 = join8(ld8(qp + 32 + half * 8), ld8(qp + 48 + half * 8));

    // --- scores: 8 waves x 8 chunks each, fully unrolled ---
#pragma unroll
    for (int cc = 0; cc < 8; ++cc) {
        const int c = wave + cc * 8;
        const int key = c * 16 + ln16;
        const __bf16* kp = k1 + ((size_t)(bb * SS + key)) * HD + h * 64;
        v16bf b0 = join8(ld8(kp + half * 8),      ld8(kp + 16 + half * 8));
        v16bf b1 = join8(ld8(kp + 32 + half * 8), ld8(kp + 48 + half * 8));
        v8f s = vzero8();
        s = wmma_bf16(a0, b0, s);
        s = wmma_bf16(a1, b1, s);
        const float km = kmask[bb * SS + key];
#pragma unroll
        for (int r = 0; r < 8; ++r) {
            float val = s[r] * 0.125f;                 // 1/sqrt(64)
            if (masked && km == 0.f) val = NEG_BIG;
            sc[r + half * 8][key] = (__bf16)val;
        }
    }
    __syncthreads();

    // --- softmax: 16 threads per row ---
    {
        const int g = t >> 4, l = t & 15;
        float m = -3.0e38f;
        for (int j = l; j < SS; j += 16) m = fmaxf(m, (float)sc[g][j]);
        red[g][l] = m;
        __syncthreads();
        float rowmax = -3.0e38f;
#pragma unroll
        for (int i = 0; i < 16; ++i) rowmax = fmaxf(rowmax, red[g][i]);
        __syncthreads();
        float ssum = 0.f;
        for (int j = l; j < SS; j += 16) {
            float e = __expf((float)sc[g][j] - rowmax);
            sc[g][j] = (__bf16)e;
            ssum += e;
        }
        red[g][l] = ssum;
        __syncthreads();
        float tot = 0.f;
#pragma unroll
        for (int i = 0; i < 16; ++i) tot += red[g][i];
        const float inv = 1.f / tot;
        for (int j = l; j < SS; j += 16)
            sc[g][j] = (__bf16)((float)sc[g][j] * inv);
    }
    __syncthreads();

    // --- PV: wave w handles keys [w*128, w*128+128), fully unrolled ---
    v8f o[4] = {vzero8(), vzero8(), vzero8(), vzero8()};
    const int k0 = wave * 128;
    const size_t vhb = ((size_t)(h * BB + bb)) * 64;
#pragma unroll
    for (int kt = 0; kt < 4; ++kt) {
        const int ks = k0 + kt * 32;
        v16bf a = join8(ld8(&sc[ln16][ks + half * 8]),          // ds_load_b128
                        ld8(&sc[ln16][ks + 16 + half * 8]));
#pragma unroll
        for (int ns = 0; ns < 4; ++ns) {
            const int f = ns * 16 + ln16;
            const __bf16* vp = v1T + (vhb + f) * SS + ks + half * 8;
            v16bf bf = join8(ld8(vp), ld8(vp + 16));            // table-order keys
            o[ns] = wmma_bf16(a, bf, o[ns]);
        }
    }
#pragma unroll
    for (int ns = 0; ns < 4; ++ns) {
        const int n = ns * 16 + ln16;
#pragma unroll
        for (int r = 0; r < 8; ++r)
            atomicAdd(&pacc[r + half * 8][n], o[ns][r]);        // ds_add_f32
    }
    __syncthreads();

    for (int i = t; i < 1024; i += 256) {
        const int m = i >> 6, n = i & 63;
        const int row = qt * 16 + m;
        float val = pacc[m][n];
        if (masked) val *= qmask[bb * SS + row];
        out[((size_t)(bb * SS + row)) * HD + h * 64 + n] = val;
    }
}

// ---------------------------------------------------------------------------
// Kernel 4: residual + LayerNorm, in-place on d_out
// ---------------------------------------------------------------------------
__global__ __launch_bounds__(256) void ln_kernel(const float* __restrict__ q,
                                                 const float* __restrict__ gamma,
                                                 const float* __restrict__ beta,
                                                 float* __restrict__ out) {
    __shared__ float rs[256], rs2[256];
    const int row = blockIdx.x;
    const int t = threadIdx.x;
    const size_t base = (size_t)row * HD;
    float x[4];
    float s = 0.f, s2 = 0.f;
#pragma unroll
    for (int i = 0; i < 4; ++i) {
        const int j = t + 256 * i;
        x[i] = out[base + j] + q[base + j];
        s += x[i];
        s2 += x[i] * x[i];
    }
    rs[t] = s; rs2[t] = s2;
    __syncthreads();
    for (int st = 128; st > 0; st >>= 1) {
        if (t < st) { rs[t] += rs[t + st]; rs2[t] += rs2[t + st]; }
        __syncthreads();
    }
    const float mean = rs[0] * (1.f / HD);
    const float var  = rs2[0] * (1.f / HD) - mean * mean;
    const float rstd = rsqrtf(var + 1e-5f);
#pragma unroll
    for (int i = 0; i < 4; ++i) {
        const int j = t + 256 * i;
        out[base + j] = (x[i] - mean) * rstd * gamma[j] + beta[j];
    }
}

// ---------------------------------------------------------------------------
extern "C" void kernel_launch(void* const* d_in, const int* in_sizes, int n_in,
                              void* d_out, int out_size, void* d_ws, size_t ws_size,
                              hipStream_t stream) {
    const float* q     = (const float*)d_in[0];
    const float* k     = (const float*)d_in[1];
    const float* v     = (const float*)d_in[2];
    const float* W     = (const float*)d_in[3];
    const float* bias  = (const float*)d_in[4];
    const float* gamma = (const float*)d_in[5];
    const float* beta  = (const float*)d_in[6];
    const int*   mask  = (const int*)d_in[7];

    const size_t ACT = (size_t)BB * SS * HD;          // 4M elements
    char* w = (char*)d_ws;
    __bf16* xq16 = (__bf16*)w;             w += ACT * 2;
    __bf16* xk16 = (__bf16*)w;             w += ACT * 2;
    __bf16* xv16 = (__bf16*)w;             w += ACT * 2;
    __bf16* W16  = (__bf16*)w;             w += (size_t)HD * HD * 2;
    __bf16* q1   = (__bf16*)w;             w += ACT * 2;
    __bf16* k1   = (__bf16*)w;             w += ACT * 2;
    __bf16* v1T  = (__bf16*)w;             w += ACT * 2;
    float*  km   = (float*)w;              w += (size_t)BB * SS * 4;
    float*  qm   = (float*)w;
    float*  out  = (float*)d_out;

    const int rows = BB * SS;                         // 4096

    cvt_kernel<<<(int)(ACT / 4 / 256), 256, 0, stream>>>(q, xq16);
    cvt_kernel<<<(int)(ACT / 4 / 256), 256, 0, stream>>>(k, xk16);
    cvt_kernel<<<(int)(ACT / 4 / 256), 256, 0, stream>>>(v, xv16);
    cvt_kernel<<<(HD * HD / 4) / 256, 256, 0, stream>>>(W, W16);

    mask_kernel<<<rows, 256, 0, stream>>>(q, k, qm, km);
    // 3 mats * 256 Mtiles * 16 Ngroups = 12288 waves / 8 waves-per-block
    qkv_proj_kernel<<<1536, 256, 0, stream>>>(xq16, xk16, xv16, W16, bias, q1, k1, v1T);
    // B * H * (S/16) = 4096 blocks
    attention_kernel<<<BB * NH * (SS / 16), 256, 0, stream>>>(q1, k1, v1T, km, qm, mask, out);
    ln_kernel<<<rows, 256, 0, stream>>>(q, gamma, beta, out);
}
